// ChannelGroupConv_11888469475660
// MI455X (gfx1250) — matrix operand order repaired and checked by reading, not compiled, not run on hardware
//
#include <hip/hip_runtime.h>

// ---------------------------------------------------------------------------
// Masked group 1x1 conv as GEMM:  Out[co,n] = sum_ci Wmask[co,ci]*X[ci,n] + b[co]
// X: [B=4][C=128][HW=262144] f32,  W: [128][128] f32 block-lower-triangular
// (16 groups x 8 ch, hidden mask gi <= go).
// bf16 split-precision (hi+lo, 3 products) WMMA, triangular K-block skipping.
// Memory-bound design: 1.07 GB HBM traffic @ 23.3 TB/s ~= 46 us floor; the
// ~65 GFLOP of bf16 WMMA work rides far below the matrix-engine roof.
// ---------------------------------------------------------------------------

typedef __attribute__((ext_vector_type(16))) __bf16 v16bf;
typedef __attribute__((ext_vector_type(8)))  float  v8f;

#define CH      128
#define HW      (512 * 512)
#define NBATCH  4
#define NTILES  ((NBATCH * HW) / 16)   // 65536 column tiles of 16 pixels

union BfPack {
    v16bf        v;
    unsigned int u[8];
};

// Packed index of surviving (rowtile r, kblock kb) weight tiles under the
// block-triangular mask: rowtile r needs kblocks 0..r/2; prefix sums give
// off[r] = {0,1,2,4,6,9,12,16}. Closed form => folds to immediates when the
// compute loops are unrolled (no constant-memory loads).
__host__ __device__ constexpr int tile_off(int r) {
    const int m = r >> 1;
    return (m + 1) * (m + (r & 1));
}

__global__ __launch_bounds__(256)
void ChannelGroupConv_wmma_bf16x3(const float* __restrict__ x,
                                  const float* __restrict__ weight,
                                  const float* __restrict__ bias,
                                  float* __restrict__ out) {
    // ------------------------------------------------------------------
    // LDS: masked weights, pre-swizzled into WMMA A-operand lane layout.
    //   A (16x32 bf16): lane l holds row M = l&15; half = l>>4;
    //   dword j holds K = (j>>2)*16 + half*8 + (j&3)*2 + {0,1}
    // Only the 20 non-zero (r,kb) tiles are stored: 2 x 20 KiB + bias.
    // ------------------------------------------------------------------
    __shared__ __align__(16) unsigned int sWhi[20][32][8];
    __shared__ __align__(16) unsigned int sWlo[20][32][8];
    __shared__ float sBias[CH];

    const int tid = threadIdx.x;

    // Build loop iterates the full 8x4 (r,kb) grid arithmetically and skips
    // masked pairs -- no lookup tables, no global constant loads.
    for (int idx = tid; idx < 8 * 4 * 32 * 8; idx += 256) {
        const int r    = idx >> 10;          // 0..7
        const int kb   = (idx >> 8) & 3;     // 0..3
        if (kb > (r >> 1)) continue;         // masked-out tile: nothing stored
        const int lane = (idx >> 3) & 31;
        const int j    = idx & 7;
        const int t    = tile_off(r) + kb;   // packed tile slot 0..19

        const int co   = 16 * r + (lane & 15);
        const int half = lane >> 4;
        const int kloc = ((j >> 2) * 16) + half * 8 + ((j & 3) * 2);
        const int ci0  = kb * 32 + kloc;
        const int ci1  = ci0 + 1;

        float w0 = weight[co * CH + ci0];
        float w1 = weight[co * CH + ci1];
        if ((ci0 >> 3) > (co >> 3)) w0 = 0.0f;   // hidden mask: gi <= go
        if ((ci1 >> 3) > (co >> 3)) w1 = 0.0f;

        const unsigned int u0 = __builtin_bit_cast(unsigned int, w0);
        const unsigned int u1 = __builtin_bit_cast(unsigned int, w1);
        sWhi[t][lane][j] = (u0 >> 16) | (u1 & 0xffff0000u);

        const float h0 = __builtin_bit_cast(float, u0 & 0xffff0000u);
        const float h1 = __builtin_bit_cast(float, u1 & 0xffff0000u);
        const unsigned int l0 = __builtin_bit_cast(unsigned int, w0 - h0);
        const unsigned int l1 = __builtin_bit_cast(unsigned int, w1 - h1);
        sWlo[t][lane][j] = (l0 >> 16) | (l1 & 0xffff0000u);
    }
    if (tid < CH) sBias[tid] = bias[tid];
    __syncthreads();

    // ------------------------------------------------------------------
    // Each wave computes a 16-pixel column tile for all 128 out channels.
    // ------------------------------------------------------------------
    const int lane = tid & 31;
    const int half = lane >> 4;
    const long long wave_id = (long long)blockIdx.x * 8 + (tid >> 5);
    const long long nwaves  = (long long)gridDim.x * 8;

    for (long long tile = wave_id; tile < NTILES; tile += nwaves) {
        const long long p0 = tile * 16;
        const int b = (int)(p0 >> 18);         // / HW (2^18)
        const int p = (int)(p0 & (HW - 1));    // tile never crosses a plane

        const float* __restrict__ xb = x   + (size_t)b * CH * HW + p + (lane & 15);
        float*       __restrict__ ob = out + (size_t)b * CH * HW + p + (lane & 15);

        v8f acc[8] = {};

#pragma unroll
        for (int kb = 0; kb < 4; ++kb) {
            // ---- load B tile (32x16 f32) and split to bf16 hi/lo in regs ----
            // B layout (matches A): dword j packs K = (j>>2)*16 + half*8 + (j&3)*2
            // (lo16) and +1 (hi16); column N = lane&15.
            BfPack bhi, blo;
#pragma unroll
            for (int j = 0; j < 8; ++j) {
                const int k0 = kb * 32 + (j >> 2) * 16 + half * 8 + (j & 3) * 2;
                const float f0 = xb[(size_t)k0 * HW];
                const float f1 = xb[(size_t)(k0 + 1) * HW];
                const unsigned int u0 = __builtin_bit_cast(unsigned int, f0);
                const unsigned int u1 = __builtin_bit_cast(unsigned int, f1);
                bhi.u[j] = (u0 >> 16) | (u1 & 0xffff0000u);
                const float h0 = __builtin_bit_cast(float, u0 & 0xffff0000u);
                const float h1 = __builtin_bit_cast(float, u1 & 0xffff0000u);
                const unsigned int l0 = __builtin_bit_cast(unsigned int, f0 - h0);
                const unsigned int l1 = __builtin_bit_cast(unsigned int, f1 - h1);
                blo.u[j] = (l0 >> 16) | (l1 & 0xffff0000u);
            }

            // ---- triangular: rowtile r uses this kblock only if r >= 2*kb ----
#pragma unroll
            for (int r = 0; r < 8; ++r) {
                if (r >= 2 * kb) {
                    const int t = tile_off(r) + kb;   // compile-time constant
                    BfPack ahi, alo;
#pragma unroll
                    for (int q = 0; q < 8; ++q) {
                        ahi.u[q] = sWhi[t][lane][q];
                        alo.u[q] = sWlo[t][lane][q];
                    }
                    // acc += Wlo*Xhi + Whi*Xlo + Whi*Xhi  (split-fp32 emulation)
                    acc[r] = __builtin_amdgcn_wmma_f32_16x16x32_bf16(
                        false, alo.v, false, bhi.v, (short)0, acc[r], false, false);
                    acc[r] = __builtin_amdgcn_wmma_f32_16x16x32_bf16(
                        false, ahi.v, false, blo.v, (short)0, acc[r], false, false);
                    acc[r] = __builtin_amdgcn_wmma_f32_16x16x32_bf16(
                        false, ahi.v, false, bhi.v, (short)0, acc[r], false, false);
                }
            }
        }

        // ---- bias + store: C/D layout: dword v -> M = v + 8*half, N = lane&15 ----
#pragma unroll
        for (int r = 0; r < 8; ++r) {
#pragma unroll
            for (int v = 0; v < 8; ++v) {
                const int co = 16 * r + v + 8 * half;
                ob[(size_t)co * HW] = acc[r][v] + sBias[co];
            }
        }
    }
}

extern "C" void kernel_launch(void* const* d_in, const int* in_sizes, int n_in,
                              void* d_out, int out_size, void* d_ws, size_t ws_size,
                              hipStream_t stream) {
    (void)in_sizes; (void)n_in; (void)d_ws; (void)ws_size; (void)out_size;
    const float* x      = (const float*)d_in[0];
    const float* weight = (const float*)d_in[1];
    const float* bias   = (const float*)d_in[2];
    float*       out    = (float*)d_out;

    // 1024 blocks x 8 waves = 8192 waves; 8 column tiles per wave (grid-stride).
    ChannelGroupConv_wmma_bf16x3<<<dim3(1024), dim3(256), 0, stream>>>(
        x, weight, bias, out);
}
